// Net_32091995636112
// MI455X (gfx1250) — compile-verified, matrix operand order checked by reading, not run on hardware
//
#include <hip/hip_runtime.h>
#include <hip/hip_bf16.h>

// ---------------- problem constants ----------------
#define NN      50000
#define EE      1600000
#define ET      1650000      // EE + NN self loops
#define IN_DIM  300
#define KPAD    320          // 300 padded to multiple of 32
#define HID     256
#define HEADS   8
#define OC      32
#define LAYERS  4
#define BQ      1024
#define NC      3
#define MPAD    50048        // ceil(50000/128)*128
#define NBLK    196          // ceil(NN/256) for the scan

typedef __attribute__((ext_vector_type(16))) _Float16 v16h;
typedef __attribute__((ext_vector_type(8)))  float    v8f;

// fp32 -> fp16 with optional zero padding (used for x, W1, conv weights)
__global__ void cvt_pad_kernel(const float* __restrict__ src, _Float16* __restrict__ dst,
                               int rs, int cs, int rd, int cd) {
  int t = blockIdx.x * blockDim.x + threadIdx.x;
  if (t >= rd * cd) return;
  int r = t / cd, c = t - r * cd;
  float v = (r < rs && c < cs) ? src[r * cs + c] : 0.0f;
  dst[t] = (_Float16)v;
}

// ---------------- WMMA GEMM: C[M,HID] = A[M,K](f16) * B[K,HID](f16) ----------------
#define BM 128
#define BN 128
#define BK 32
#define LDA_T 32   // halfs
#define LDB_T 40   // halfs (padded to dodge bank conflicts, keeps 16B alignment)

__device__ __forceinline__ v16h load_frag(const _Float16* p, int g0) {
  // fragment = halfs [g0..g0+7] and [g0+16..g0+23] of one 32-half row
  union { uint4 q[2]; v16h f; } u;
  u.q[0] = *(const uint4*)(p + g0);
  u.q[1] = *(const uint4*)(p + g0 + 16);
  return u.f;
}

__global__ __launch_bounds__(256) void gemm_f16_wmma(
    const _Float16* __restrict__ A, const _Float16* __restrict__ B,
    float* __restrict__ C, int K, int lda, int ldb, int ldc) {
  __shared__ _Float16 At[BM * LDA_T];   // 8 KB
  __shared__ _Float16 Bt[BN * LDB_T];   // 10 KB (n-major, transposed)

  const int tid  = threadIdx.x;
  const int lane = tid & 31;
  const int wave = tid >> 5;
  const int wm   = wave & 3;    // 4 strips of 32 rows
  const int wn   = wave >> 2;   // 2 strips of 64 cols
  const int bm   = blockIdx.y * BM;
  const int bn   = blockIdx.x * BN;
  const int g0   = (lane < 16) ? 0 : 8;

  v8f acc[2][4];
  for (int i = 0; i < 2; ++i)
    for (int j = 0; j < 4; ++j)
      for (int r = 0; r < 8; ++r) acc[i][j][r] = 0.0f;

  const int arow = tid >> 1;            // 0..127
  const int acol = (tid & 1) * 16;      // 0 / 16
  const int bk   = tid >> 3;            // 0..31
  const int bnb  = (tid & 7) * 16;      // 0..112

  for (int k0 = 0; k0 < K; k0 += BK) {
    // stage A tile (row-major, 32 halfs/row)
    {
      const _Float16* gp = A + (size_t)(bm + arow) * lda + k0 + acol;
      uint4 v0 = ((const uint4*)gp)[0];
      uint4 v1 = ((const uint4*)gp)[1];
      uint4* sp = (uint4*)(At + arow * LDA_T + acol);
      sp[0] = v0; sp[1] = v1;
      if (k0 + BK < K) __builtin_prefetch(gp + BK, 0, 3);   // global_prefetch
    }
    // stage B tile transposed (n-major)
    {
      const _Float16* gp = B + (size_t)(k0 + bk) * ldb + bn + bnb;
      union { uint4 q[2]; _Float16 h[16]; } u;
      u.q[0] = ((const uint4*)gp)[0];
      u.q[1] = ((const uint4*)gp)[1];
      if (k0 + BK < K) __builtin_prefetch(gp + (size_t)BK * ldb, 0, 3);
      for (int j = 0; j < 16; ++j) Bt[(bnb + j) * LDB_T + bk] = u.h[j];
    }
    __syncthreads();

    v16h af[2], bf[4];
    for (int i = 0; i < 2; ++i)
      af[i] = load_frag(At + (wm * 32 + i * 16 + (lane & 15)) * LDA_T, g0);
    for (int j = 0; j < 4; ++j)
      bf[j] = load_frag(Bt + (wn * 64 + j * 16 + (lane & 15)) * LDB_T, g0);

    for (int i = 0; i < 2; ++i)
      for (int j = 0; j < 4; ++j)
        acc[i][j] = __builtin_amdgcn_wmma_f32_16x16x32_f16(
            false, af[i], false, bf[j], (short)0, acc[i][j], false, false);
    __syncthreads();
  }

  // store: C/D layout -> lane n = lane&15, rows mbase..mbase+7
  const int mbase = (lane < 16) ? 0 : 8;
  const int ncol  = lane & 15;
  for (int i = 0; i < 2; ++i)
    for (int j = 0; j < 4; ++j) {
      int r0 = bm + wm * 32 + i * 16 + mbase;
      int c0 = bn + wn * 64 + j * 16 + ncol;
      for (int r = 0; r < 8; ++r)
        C[(size_t)(r0 + r) * ldc + c0] = acc[i][j][r];
    }
}

// ---------------- bias epilogue for lin1; writes f32 and f16 ----------------
__global__ __launch_bounds__(256) void bias_kernel(
    const float* __restrict__ in, const float* __restrict__ bias,
    float* __restrict__ hf, _Float16* __restrict__ hh) {
  int n = blockIdx.x;            // 0..MPAD-1
  int c = threadIdx.x;           // 0..255
  size_t t = (size_t)n * HID + c;
  float v = 0.0f;
  if (n < NN) v = in[t] + bias[c];
  hf[t] = v;
  hh[t] = (_Float16)v;
}

// ---------------- CSR construction (once per call; edges shared by all layers) ----
__global__ void count_kernel(const int* __restrict__ etgt, int* __restrict__ cnt) {
  int e = blockIdx.x * blockDim.x + threadIdx.x;
  if (e >= ET) return;
  int d = (e < EE) ? etgt[e] : (e - EE);
  atomicAdd(cnt + d, 1);
}

__global__ __launch_bounds__(256) void scan_block_kernel(
    const int* __restrict__ cnt, int* __restrict__ rowptr, int* __restrict__ bsum) {
  __shared__ int s[256];
  int t = threadIdx.x;
  int g = blockIdx.x * 256 + t;
  int v = (g < NN) ? cnt[g] : 0;
  s[t] = v;
  __syncthreads();
  for (int off = 1; off < 256; off <<= 1) {
    int x = (t >= off) ? s[t - off] : 0;
    __syncthreads();
    s[t] += x;
    __syncthreads();
  }
  if (g < NN) rowptr[g] = s[t] - v;          // exclusive within block
  if (t == 255) bsum[blockIdx.x] = s[255];   // block total
}

__global__ __launch_bounds__(256) void scan_totals_kernel(
    const int* __restrict__ bsum, int* __restrict__ boff, int nb) {
  __shared__ int s[256];
  int t = threadIdx.x;
  int v = (t < nb) ? bsum[t] : 0;
  s[t] = v;
  __syncthreads();
  for (int off = 1; off < 256; off <<= 1) {
    int x = (t >= off) ? s[t - off] : 0;
    __syncthreads();
    s[t] += x;
    __syncthreads();
  }
  if (t < nb) boff[t] = s[t] - v;            // exclusive
}

__global__ void add_off_kernel(int* __restrict__ rowptr, const int* __restrict__ boff) {
  int g = blockIdx.x * blockDim.x + threadIdx.x;
  if (g < NN) rowptr[g] += boff[g >> 8];
  if (g == 0) rowptr[NN] = ET;               // total edge count is a constant
}

__global__ void fill_kernel(const int* __restrict__ esrc, const int* __restrict__ etgt,
                            int* __restrict__ cursor, int* __restrict__ col) {
  int e = blockIdx.x * blockDim.x + threadIdx.x;
  if (e >= ET) return;
  int s, d;
  if (e < EE) { s = esrc[e]; d = etgt[e]; } else { s = d = e - EE; }
  int pos = atomicAdd(cursor + d, 1);
  col[pos] = s;
}

// ---------------- per-node attention scores ----------------
__global__ void scores_kernel(const float* __restrict__ hw,
                              const float* __restrict__ attd,
                              const float* __restrict__ atts,
                              float* __restrict__ sd, float* __restrict__ ss) {
  int t = blockIdx.x * blockDim.x + threadIdx.x;
  if (t >= NN * HEADS) return;
  int n = t >> 3, h = t & 7;
  const float* p  = hw + (size_t)n * HID + h * OC;
  const float* ad = attd + h * OC;
  const float* as = atts + h * OC;
  float a = 0.0f, b = 0.0f;
  for (int o = 0; o < OC; ++o) { float v = p[o]; a += v * ad[o]; b += v * as[o]; }
  sd[t] = a; ss[t] = b;
}

// ---------------- fused GAT layer: softmax + aggregate + bias + ELU, no atomics ----
// 1 block = 1 target node; thread = output channel c; head = c>>5 (wave-uniform).
__global__ __launch_bounds__(256) void gat_fused_kernel(
    const int* __restrict__ rowptr, const int* __restrict__ col,
    const float* __restrict__ hw, const float* __restrict__ sd,
    const float* __restrict__ ss, const float* __restrict__ bias,
    float* __restrict__ hf, _Float16* __restrict__ hb) {
  __shared__ float red[8][8];   // [wave][head] partial maxima
  int n = blockIdx.x;
  int c = threadIdx.x;
  size_t tout = (size_t)n * HID + c;
  if (n >= NN) { hf[tout] = 0.0f; hb[tout] = (_Float16)0.0f; return; }

  const int lane = c & 31;
  const int wave = c >> 5;
  const int h    = c >> 5;          // head index
  const int beg  = rowptr[n];
  const int end  = rowptr[n + 1];   // deg >= 1 (self loop)

  float sdn[HEADS];
  for (int q = 0; q < HEADS; ++q) sdn[q] = sd[n * HEADS + q];

  // ---- phase 1: per-head max of leaky_relu(s_dst[n] + s_src[src]) ----
  float mx[HEADS];
  for (int q = 0; q < HEADS; ++q) mx[q] = -3.4e38f;
  for (int i = beg + c; i < end; i += 256) {
    int s = col[i];
    const float* sp = ss + (size_t)s * HEADS;
    for (int q = 0; q < HEADS; ++q) {
      float a = sdn[q] + sp[q];
      a = (a > 0.0f) ? a : 0.2f * a;
      mx[q] = fmaxf(mx[q], a);
    }
  }
  for (int q = 0; q < HEADS; ++q)
    for (int o = 16; o >= 1; o >>= 1)
      mx[q] = fmaxf(mx[q], __shfl_xor(mx[q], o, 32));
  if (lane == 0)
    for (int q = 0; q < HEADS; ++q) red[wave][q] = mx[q];
  __syncthreads();
  float mh = -3.4e38f;
  for (int w = 0; w < 8; ++w) mh = fmaxf(mh, red[w][h]);

  // ---- phase 2: weighted gather-accumulate + denominator (identical per lane) ----
  const float sdh = sdn[h];
  float acc = 0.0f, den = 0.0f;
  int i = beg;
  for (; i + 1 < end; i += 2) {            // 2-edge unroll for load ILP
    int s0 = col[i], s1 = col[i + 1];
    float a0 = sdh + ss[(size_t)s0 * HEADS + h];
    float a1 = sdh + ss[(size_t)s1 * HEADS + h];
    a0 = (a0 > 0.0f) ? a0 : 0.2f * a0;
    a1 = (a1 > 0.0f) ? a1 : 0.2f * a1;
    float w0 = __expf(a0 - mh);
    float w1 = __expf(a1 - mh);
    float f0 = hw[(size_t)s0 * HID + c];
    float f1 = hw[(size_t)s1 * HID + c];
    den += w0 + w1;
    acc += w0 * f0 + w1 * f1;
  }
  if (i < end) {
    int s0 = col[i];
    float a0 = sdh + ss[(size_t)s0 * HEADS + h];
    a0 = (a0 > 0.0f) ? a0 : 0.2f * a0;
    float w0 = __expf(a0 - mh);
    den += w0;
    acc += w0 * hw[(size_t)s0 * HID + c];
  }

  float v = acc / (den + 1e-16f) + bias[c];
  v = (v > 0.0f) ? v : (__expf(v) - 1.0f);   // ELU
  hf[tout] = v;
  hb[tout] = (_Float16)v;
}

// ---------------- final classifier head ----------------
__global__ void head_kernel(const float* __restrict__ hf, const int* __restrict__ mask,
                            const float* __restrict__ W3, const float* __restrict__ b3,
                            float* __restrict__ out) {
  int t = blockIdx.x * blockDim.x + threadIdx.x;
  if (t >= BQ * NC) return;
  int b = t / NC, cls = t - b * NC;
  int n0 = mask[b * 2], n1 = mask[b * 2 + 1];
  const float* h0 = hf + (size_t)n0 * HID;
  const float* h1 = hf + (size_t)n1 * HID;
  float acc = b3[cls];
  for (int j = 0; j < HID; ++j) acc += h0[j] * W3[j * NC + cls];
  for (int j = 0; j < HID; ++j) acc += h1[j] * W3[(HID + j) * NC + cls];
  out[t] = acc;
}

// ---------------- host side ----------------
extern "C" void kernel_launch(void* const* d_in, const int* in_sizes, int n_in,
                              void* d_out, int out_size, void* d_ws, size_t ws_size,
                              hipStream_t stream) {
  const float* x        = (const float*)d_in[0];
  const int*   eidx     = (const int*)d_in[1];       // [2, E]
  const int*   tmask    = (const int*)d_in[2];       // [B, 2]
  const float* lin1_W   = (const float*)d_in[3];     // [300,256]
  const float* lin1_b   = (const float*)d_in[4];
  const float* convs_W  = (const float*)d_in[5];     // [4,256,256]
  const float* att_dst  = (const float*)d_in[6];     // [4,8,32]
  const float* att_src  = (const float*)d_in[7];
  const float* convs_b  = (const float*)d_in[8];     // [4,256]
  const float* lin3_W   = (const float*)d_in[9];     // [512,3]
  const float* lin3_b   = (const float*)d_in[10];
  float* out = (float*)d_out;

  const int* esrc = eidx;
  const int* etgt = eidx + EE;

  // workspace carve-out (256B aligned)
  char* ws = (char*)d_ws;
  size_t off = 0;
  auto carve = [&](size_t bytes) -> char* {
    off = (off + 255) & ~(size_t)255;
    char* p = ws + off;
    off += bytes;
    return p;
  };
  _Float16* xb     = (_Float16*)carve((size_t)MPAD * KPAD * 2);
  _Float16* w1b    = (_Float16*)carve((size_t)KPAD * HID * 2);
  _Float16* wlb    = (_Float16*)carve((size_t)LAYERS * HID * HID * 2);
  _Float16* hb     = (_Float16*)carve((size_t)MPAD * HID * 2);
  float*    hf     = (float*)carve((size_t)MPAD * HID * 4);
  float*    hw     = (float*)carve((size_t)MPAD * HID * 4);
  float*    sd     = (float*)carve((size_t)NN * HEADS * 4);
  float*    ssn    = (float*)carve((size_t)NN * HEADS * 4);
  int*      cnt    = (int*)carve((size_t)NN * 4);
  int*      rowptr = (int*)carve((size_t)(NN + 1) * 4);
  int*      cursor = (int*)carve((size_t)NN * 4);
  int*      bsum   = (int*)carve(256 * 4);
  int*      boff   = (int*)carve(256 * 4);
  int*      col    = (int*)carve((size_t)ET * 4);
  (void)ws_size; (void)in_sizes; (void)n_in; (void)out_size;

  // 1) convert + pad inputs to f16
  {
    int tot = MPAD * KPAD;
    cvt_pad_kernel<<<(tot + 255) / 256, 256, 0, stream>>>(x, xb, NN, IN_DIM, MPAD, KPAD);
    tot = KPAD * HID;
    cvt_pad_kernel<<<(tot + 255) / 256, 256, 0, stream>>>(lin1_W, w1b, IN_DIM, HID, KPAD, HID);
    tot = LAYERS * HID * HID;
    cvt_pad_kernel<<<(tot + 255) / 256, 256, 0, stream>>>(convs_W, wlb, LAYERS * HID, HID,
                                                          LAYERS * HID, HID);
  }

  // 2) CSR build (shared by all 4 layers)
  hipMemsetAsync(cnt, 0, (size_t)NN * 4, stream);
  count_kernel<<<(ET + 255) / 256, 256, 0, stream>>>(etgt, cnt);
  scan_block_kernel<<<NBLK, 256, 0, stream>>>(cnt, rowptr, bsum);
  scan_totals_kernel<<<1, 256, 0, stream>>>(bsum, boff, NBLK);
  add_off_kernel<<<(NN + 255) / 256, 256, 0, stream>>>(rowptr, boff);
  hipMemcpyAsync(cursor, rowptr, (size_t)NN * 4, hipMemcpyDeviceToDevice, stream);
  fill_kernel<<<(ET + 255) / 256, 256, 0, stream>>>(esrc, etgt, cursor, col);

  dim3 ggrid(HID / BN, MPAD / BM);   // (2, 391)

  // 3) h = x @ lin1_W + lin1_b
  gemm_f16_wmma<<<ggrid, 256, 0, stream>>>(xb, w1b, hw, KPAD, KPAD, HID, HID);
  bias_kernel<<<MPAD, 256, 0, stream>>>(hw, lin1_b, hf, hb);

  // 4) GAT layers: GEMM -> scores -> fused softmax/aggregate/bias/ELU
  for (int l = 0; l < LAYERS; ++l) {
    gemm_f16_wmma<<<ggrid, 256, 0, stream>>>(hb, wlb + (size_t)l * HID * HID, hw,
                                             HID, HID, HID, HID);
    scores_kernel<<<(NN * HEADS + 255) / 256, 256, 0, stream>>>(
        hw, att_dst + (size_t)l * HEADS * OC, att_src + (size_t)l * HEADS * OC, sd, ssn);
    gat_fused_kernel<<<MPAD, 256, 0, stream>>>(rowptr, col, hw, sd, ssn,
                                               convs_b + (size_t)l * HID, hf, hb);
  }

  // 5) head
  head_kernel<<<(BQ * NC + 255) / 256, 256, 0, stream>>>(hf, tmask, lin3_W, lin3_b, out);
}
